// QuantLinear_16003048145696
// MI455X (gfx1250) — matrix operand order repaired and checked by reading, compiled, not run on hardware
//
#include <hip/hip_runtime.h>
#include <hip/hip_bf16.h>
#include <cstdint>

// ---------------------------------------------------------------------------
// QuantLinear for MI455X (gfx1250, wave32, WMMA).
//   Phase 1: dequant 4-bit e2m2 + sign bitmap -> bf16 W[K][N] in workspace
//            (90 MB; L2-resident for phase 2, L2 = 192 MB).
//   Phase 2: 128x128x64-tiled bf16 GEMM, v_wmma_f32_16x16x32_bf16,
//            register-staged LDS double buffering (1 barrier/iter),
//            global_prefetch for the x stream, scale+bias in f32 epilogue.
// ---------------------------------------------------------------------------

#define K_DIM 4096
#define N_DIM 11008

typedef __attribute__((ext_vector_type(8)))  float   v8f;
typedef __attribute__((ext_vector_type(8)))  __bf16  v8bf;
typedef __attribute__((ext_vector_type(16))) __bf16  v16bf;

// e2m2 magnitude -> bf16 bits (all values exact in bf16).
// idx = e[1:0]m[1:0]: e>=1 -> (4+m) * 2^(e-2) ; e==0 -> m * 0.5
__device__ __forceinline__ unsigned mag_bf16_bits(int idx) {
    int e = idx >> 2, m = idx & 3;
    if (e) return (unsigned)(((127 + e) << 7) | (m << 5));
    if (!m) return 0u;
    return (unsigned)(((126 + (m >> 1)) << 7) | ((m == 3) << 6));
}

__device__ __forceinline__ unsigned f32_to_bf16_bits(float f) {
    __bf16 b = (__bf16)f;
    return (unsigned)__builtin_bit_cast(unsigned short, b);
}

// ---------------------------------------------------------------------------
// Phase 1: dequant. grid = (N/256, K/8), block = 128. Two consecutive n per
// thread -> every store is a coalesced b32, loads are b64.
// ---------------------------------------------------------------------------
__global__ __launch_bounds__(128) void dequant_bf16_kernel(
    const int* __restrict__ qw, const int* __restrict__ sm,
    unsigned short* __restrict__ w) {
    const int n  = (blockIdx.x * 128 + threadIdx.x) * 2;  // even n
    const int kb = blockIdx.y;                            // 0..K/8-1
    const int2 qv = *(const int2*)&qw[(size_t)kb * N_DIM + n];
    const int2 sv = *(const int2*)&sm[(size_t)(kb >> 2) * N_DIM + n];
    const int sbase = (kb & 3) * 8;
#pragma unroll
    for (int j = 0; j < 8; ++j) {
        unsigned b0 = mag_bf16_bits((qv.x >> (4 * j)) & 15) |
                      (((unsigned)(sv.x >> (sbase + j)) & 1u) << 15);
        unsigned b1 = mag_bf16_bits((qv.y >> (4 * j)) & 15) |
                      (((unsigned)(sv.y >> (sbase + j)) & 1u) << 15);
        *(unsigned*)&w[(size_t)(kb * 8 + j) * N_DIM + n] = b0 | (b1 << 16);
    }
}

// ---------------------------------------------------------------------------
// Phase 2: GEMM. 256 threads / 8 waves, block tile 128(M) x 128(N), BK=64.
// Wave (wm,wn): 64(M) x 32(N) region = 4x2 grid of 16x16 WMMA tiles.
// LDS: double-buffered A row-major [128][80] + B col-major [128][80] bf16.
// ---------------------------------------------------------------------------
template <bool FUSED>
__global__ __launch_bounds__(256) void qlin_gemm_kernel(
    const float* __restrict__ x, const unsigned short* __restrict__ wbf,
    const int* __restrict__ qw, const int* __restrict__ sm,
    const float* __restrict__ scales, const float* __restrict__ bias,
    float* __restrict__ out) {

    __shared__ unsigned short Abuf[2][128 * 80];   // [m][k], stride 80 (160 B)
    __shared__ unsigned short Bbuf[2][128 * 80];   // [n][k], stride 80 (160 B)

    const int tid  = threadIdx.x;
    const int lane = tid & 31;
    const int wave = tid >> 5;
    const int wm   = wave & 1;    // 0..1 : M half (64 rows)
    const int wn   = wave >> 1;   // 0..3 : N quarter (32 cols)

    const int bm = blockIdx.y * 128;
    const int bn = blockIdx.x * 128;

    v8f acc[4][2];
#pragma unroll
    for (int mt = 0; mt < 4; ++mt)
#pragma unroll
        for (int nt = 0; nt < 2; ++nt) acc[mt][nt] = (v8f)0.0f;

    const int half = lane >> 4;     // 0/1
    const int l16  = lane & 15;

    // ---- register staging for the next tile ----
    float4 ax[8];     // A: x[128][64] f32, 8 float4 per thread
    uint4  bwreg[4];  // B (!FUSED): raw bf16 chunks
    int    fq[4];     // B (FUSED): qweight words
    int    fs = 0;    // B (FUSED): sign word

    auto load_tile = [&](int kt) {
        const int c4 = tid & 15;                    // float4 column
        const int r0 = tid >> 4;                    // 0..15
        const float* xp = x + (size_t)(bm + r0) * K_DIM + kt + c4 * 4;
#pragma unroll
        for (int r = 0; r < 8; ++r)
            ax[r] = *(const float4*)(xp + (size_t)r * 16 * K_DIM);
        if (!FUSED) {
            const int nc = tid & 15, kr = tid >> 4;
#pragma unroll
            for (int r = 0; r < 4; ++r)
                bwreg[r] = *(const uint4*)(wbf + (size_t)(kt + kr + r * 16) * N_DIM +
                                           bn + nc * 8);
        } else {
            const int n_l = tid & 127, g = tid >> 7;
            fs = sm[(size_t)((kt >> 5) + g) * N_DIM + bn + n_l];
#pragma unroll
            for (int i = 0; i < 4; ++i)
                fq[i] = qw[(size_t)((kt >> 3) + g * 4 + i) * N_DIM + bn + n_l];
        }
    };

    auto store_tile = [&](int p) {
        const int c4 = tid & 15;
        const int r0 = tid >> 4;
#pragma unroll
        for (int r = 0; r < 8; ++r) {
            unsigned p0 = f32_to_bf16_bits(ax[r].x) | (f32_to_bf16_bits(ax[r].y) << 16);
            unsigned p1 = f32_to_bf16_bits(ax[r].z) | (f32_to_bf16_bits(ax[r].w) << 16);
            unsigned* dst = (unsigned*)&Abuf[p][(r0 + r * 16) * 80 + c4 * 4];
            dst[0] = p0; dst[1] = p1;
        }
        if (!FUSED) {
            const int nc = tid & 15, kr = tid >> 4;
#pragma unroll
            for (int r = 0; r < 4; ++r) {
                const int kk = kr + r * 16;
                const int nbase = nc * 8;
                uint4 d = bwreg[r];
                Bbuf[p][(nbase + 0) * 80 + kk] = (unsigned short)(d.x);
                Bbuf[p][(nbase + 1) * 80 + kk] = (unsigned short)(d.x >> 16);
                Bbuf[p][(nbase + 2) * 80 + kk] = (unsigned short)(d.y);
                Bbuf[p][(nbase + 3) * 80 + kk] = (unsigned short)(d.y >> 16);
                Bbuf[p][(nbase + 4) * 80 + kk] = (unsigned short)(d.z);
                Bbuf[p][(nbase + 5) * 80 + kk] = (unsigned short)(d.z >> 16);
                Bbuf[p][(nbase + 6) * 80 + kk] = (unsigned short)(d.w);
                Bbuf[p][(nbase + 7) * 80 + kk] = (unsigned short)(d.w >> 16);
            }
        } else {
            const int n_l = tid & 127, g = tid >> 7;
#pragma unroll
            for (int i = 0; i < 4; ++i) {
                unsigned h[8];
#pragma unroll
                for (int j = 0; j < 8; ++j)
                    h[j] = mag_bf16_bits((fq[i] >> (4 * j)) & 15) |
                           (((unsigned)(fs >> (i * 8 + j)) & 1u) << 15);
                uint4 pk = make_uint4(h[0] | (h[1] << 16), h[2] | (h[3] << 16),
                                      h[4] | (h[5] << 16), h[6] | (h[7] << 16));
                *(uint4*)&Bbuf[p][n_l * 80 + (g * 4 + i) * 8] = pk;
            }
        }
    };

    // ---- pipeline prologue ----
    load_tile(0);
    store_tile(0);
    __syncthreads();

    int p = 0;
    for (int kt = 0; kt < K_DIM; kt += 64) {
        const bool last = (kt + 64 >= K_DIM);
        if (!last) load_tile(kt + 64);                 // global loads in flight
        if (kt + 128 < K_DIM) {                        // prefetch x two tiles ahead
            const int prow = tid >> 1;
            const int pcol = (tid & 1) * 32;
            __builtin_prefetch(x + (size_t)(bm + prow) * K_DIM + kt + 128 + pcol, 0, 3);
        }

        // ---- compute current buffer: 2 k-steps of 32 ----
#pragma unroll
        for (int ks = 0; ks < 64; ks += 32) {
            v16bf af[4], bf[2];
#pragma unroll
            for (int mt = 0; mt < 4; ++mt) {
                // A 16x32 frag: lane holds K = ks+8*half..+7 (regs 0-3) and
                // K = ks+16+8*half..+7 (regs 4-7).
                const int row = wm * 64 + mt * 16 + l16;
                const int off = row * 80 + ks + half * 8;
                v8bf lo = *(const v8bf*)&Abuf[p][off];
                v8bf hi = *(const v8bf*)&Abuf[p][off + 16];
                af[mt] = __builtin_shufflevector(lo, hi, 0, 1, 2, 3, 4, 5, 6, 7,
                                                 8, 9, 10, 11, 12, 13, 14, 15);
            }
#pragma unroll
            for (int nt = 0; nt < 2; ++nt) {
                // B 32x16 frag: lane(n=l16) holds K = ks+8*half..+7 and
                // K = ks+16+8*half..+7.
                const int col = wn * 32 + nt * 16 + l16;
                const int off = col * 80 + ks + half * 8;
                v8bf lo = *(const v8bf*)&Bbuf[p][off];
                v8bf hi = *(const v8bf*)&Bbuf[p][off + 16];
                bf[nt] = __builtin_shufflevector(lo, hi, 0, 1, 2, 3, 4, 5, 6, 7,
                                                 8, 9, 10, 11, 12, 13, 14, 15);
            }
#pragma unroll
            for (int mt = 0; mt < 4; ++mt)
#pragma unroll
                for (int nt = 0; nt < 2; ++nt)
                    acc[mt][nt] = __builtin_amdgcn_wmma_f32_16x16x32_bf16(
                        false, af[mt], false, bf[nt], (short)0, acc[mt][nt],
                        false, false);
        }

        if (!last) {
            store_tile(p ^ 1);    // other buffer: prev reads fenced by last barrier
            __syncthreads();
        }
        p ^= 1;
    }

    // ---- epilogue: out = acc * scale[n] + bias[n] ----
#pragma unroll
    for (int nt = 0; nt < 2; ++nt) {
        const int col = bn + wn * 32 + nt * 16 + l16;
        const float sc = scales[col];
        const float bs = bias[col];
#pragma unroll
        for (int mt = 0; mt < 4; ++mt) {
            const int rbase = bm + wm * 64 + mt * 16 + half * 8;  // C: M = v + 8*half
#pragma unroll
            for (int v = 0; v < 8; ++v) {
                out[(size_t)(rbase + v) * N_DIM + col] = acc[mt][nt][v] * sc + bs;
            }
        }
    }
}

// ---------------------------------------------------------------------------
extern "C" void kernel_launch(void* const* d_in, const int* in_sizes, int n_in,
                              void* d_out, int out_size, void* d_ws, size_t ws_size,
                              hipStream_t stream) {
    const float* x      = (const float*)d_in[0];
    const int*   qw     = (const int*)d_in[1];
    const int*   sm     = (const int*)d_in[2];
    const float* scales = (const float*)d_in[3];
    const float* bias   = (const float*)d_in[4];
    float*       out    = (float*)d_out;

    const int Mtok = in_sizes[0] / K_DIM;           // 8192 tokens
    const size_t wbytes = (size_t)K_DIM * N_DIM * sizeof(unsigned short);

    dim3 gblk(N_DIM / 128, Mtok / 128);             // (86, 64)

    if (ws_size >= wbytes) {
        dequant_bf16_kernel<<<dim3(N_DIM / 256, K_DIM / 8), 128, 0, stream>>>(
            qw, sm, (unsigned short*)d_ws);
        qlin_gemm_kernel<false><<<gblk, 256, 0, stream>>>(
            x, (const unsigned short*)d_ws, qw, sm, scales, bias, out);
    } else {
        qlin_gemm_kernel<true><<<gblk, 256, 0, stream>>>(
            x, nullptr, qw, sm, scales, bias, out);
    }
}
